// TopologyConstrainedDecoder_71811853189764
// MI455X (gfx1250) — compile-verified, matrix operand order checked by reading, not run on hardware
//
#include <hip/hip_runtime.h>
#include <hip/hip_bf16.h>

typedef __attribute__((ext_vector_type(16))) _Float16 v16h;
typedef __attribute__((ext_vector_type(8)))  float    v8f;

#define HID 512
#define B   64
#define T   64
#define P   128
#define RNN_LD 1026          // row length of gru_w_ih
#define H3  (3 * HID)        // 1536

// ---------------------------------------------------------------------------
// WMMA tile loader: 16x32 f16 tile from a row-major matrix (ld = row length).
// ISA 7.12.2 16-bit A layout: lanes 0-15 hold row M=lane, K in {0..7,16..23};
// lanes 16-31 hold row M=lane-16, K in {8..15,24..31}. B (32x16, from W row-
// major N x K, since out = x @ W^T) uses the symmetric layout with N=lane&15.
// ---------------------------------------------------------------------------
__device__ __forceinline__ v16h load_tile_f16(const _Float16* __restrict__ base,
                                              int ld, int row0, int k0, int lane) {
    int r  = lane & 15;
    int kb = (lane >> 4) << 3;                    // 0 or 8
    const _Float16* p = base + (size_t)(row0 + r) * ld + k0;
    v16h t;
#pragma unroll
    for (int i = 0; i < 8; ++i) t[i] = p[kb + i];
#pragma unroll
    for (int i = 0; i < 8; ++i) t[8 + i] = p[16 + kb + i];
    return t;
}

// ---------------------------------------------------------------------------
// f16 WMMA GEMM: out[m][n] = sum_k A[m][k] * W[n][k]  (i.e. A @ W^T)
//   + bias[n] + add[m][n] + rowscale[m]*colvec[n], optional ReLU.
// One wave -> 16 x (16*NT) output. k-unrolled by 2 with ping-pong register
// double-buffering of BOTH A and B: next-chunk loads are issued before the
// current chunk's wmma group, so each wmma waits on loads issued a full
// 32-k chunk earlier (load latency hides under prior XDL work, no register
// copies between buffers). Requires K % 32 == 0 (true for all call sites).
// Block = 4 waves tiling N. grid = (N/(64*NT), M/16).
// ---------------------------------------------------------------------------
template<int NT>
__global__ void gemm_wmma_f16(const _Float16* __restrict__ A, int lda,
                              const _Float16* __restrict__ W, int ldw,
                              float* __restrict__ out, int ldo,
                              int M, int N, int K,
                              const float* __restrict__ bias,
                              const float* __restrict__ add,      // M x ldo
                              const float* __restrict__ rowscale, // M
                              const float* __restrict__ colvec,   // N
                              int relu)
{
    int lane = threadIdx.x & 31;
    int wave = threadIdx.x >> 5;
    int col0 = (blockIdx.x * 4 + wave) * (16 * NT);
    int row0 = blockIdx.y * 16;
    if (col0 >= N || row0 >= M) return;

    v8f acc[NT];
#pragma unroll
    for (int j = 0; j < NT; ++j) acc[j] = v8f{};

    // preload chunk 0
    v16h a0 = load_tile_f16(A, lda, row0, 0, lane);
    v16h b0[NT];
#pragma unroll
    for (int j = 0; j < NT; ++j) b0[j] = load_tile_f16(W, ldw, col0 + 16 * j, 0, lane);

    for (int k0 = 0; k0 < K; k0 += 64) {
        // issue loads for chunk k0+32 before consuming chunk k0
        v16h a1, b1[NT];
        if (k0 + 32 < K) {
            __builtin_prefetch(W + (size_t)(col0 + (lane & 15)) * ldw + k0 + 96, 0, 3);
            a1 = load_tile_f16(A, lda, row0, k0 + 32, lane);
#pragma unroll
            for (int j = 0; j < NT; ++j)
                b1[j] = load_tile_f16(W, ldw, col0 + 16 * j, k0 + 32, lane);
        }
#pragma unroll
        for (int j = 0; j < NT; ++j)
            acc[j] = __builtin_amdgcn_wmma_f32_16x16x32_f16(false, a0, false, b0[j],
                                                            (short)0, acc[j], false, false);
        // issue loads for chunk k0+64 before consuming chunk k0+32
        if (k0 + 64 < K) {
            a0 = load_tile_f16(A, lda, row0, k0 + 64, lane);
#pragma unroll
            for (int j = 0; j < NT; ++j)
                b0[j] = load_tile_f16(W, ldw, col0 + 16 * j, k0 + 64, lane);
        }
        if (k0 + 32 < K) {
#pragma unroll
            for (int j = 0; j < NT; ++j)
                acc[j] = __builtin_amdgcn_wmma_f32_16x16x32_f16(false, a1, false, b1[j],
                                                                (short)0, acc[j], false, false);
        }
    }

    // C/D layout: N = col + (lane&15); VGPR r -> M = r (lanes 0-15) / 8+r.
    int mb = row0 + ((lane < 16) ? 0 : 8);
#pragma unroll
    for (int j = 0; j < NT; ++j) {
        int n  = col0 + 16 * j + (lane & 15);
        float bn = bias   ? bias[n]   : 0.0f;
        float cv = colvec ? colvec[n] : 0.0f;
#pragma unroll
        for (int r = 0; r < 8; ++r) {
            int m   = mb + r;
            float v = acc[j][r] + bn;
            if (add)      v += add[(size_t)m * ldo + n];
            if (rowscale) v += rowscale[m] * cv;
            if (relu)     v  = fmaxf(v, 0.0f);
            out[(size_t)m * ldo + n] = v;
        }
    }
}

// --------------------------- setup / conversion ----------------------------
__global__ void extract_f16(_Float16* dst, const float* src,
                            int rows, int cols, int ld, int col0) {
    long i = blockIdx.x * (long)blockDim.x + threadIdx.x;
    if (i >= (long)rows * cols) return;
    int r = (int)(i / cols), c = (int)(i % cols);
    dst[i] = (_Float16)src[(long)r * ld + col0 + c];
}

__global__ void extract_col_f32(float* dst, const float* src, int rows, int ld, int col) {
    int r = blockIdx.x * blockDim.x + threadIdx.x;
    if (r < rows) dst[r] = src[(long)r * ld + col];
}

__global__ void gather_rows_f16(_Float16* dst, const float* emb, const int* ids, int nrows) {
    long i = blockIdx.x * (long)blockDim.x + threadIdx.x;
    if (i >= (long)nrows * HID) return;
    int r = (int)(i / HID), c = (int)(i % HID);
    dst[i] = (_Float16)emb[(long)ids[r] * HID + c];
}

__global__ void gather_path_f16(_Float16* dst, const float* emb, const int* path) {
    // dst row = b*P + p ; id = path[p*B + b]   (path is (P,B))
    long i = blockIdx.x * (long)blockDim.x + threadIdx.x;
    if (i >= (long)B * P * HID) return;
    int  c   = (int)(i % HID);
    long row = i / HID;
    int  p   = (int)(row % P), b = (int)(row / P);
    dst[i] = (_Float16)emb[(long)path[p * B + b] * HID + c];
}

__global__ void init_outputs_h(float* out_id, float* out_rate,
                               float* h, _Float16* h_h, const float* hidden) {
    int i = blockIdx.x * blockDim.x + threadIdx.x;   // covers 32768
    if (i < B * P) out_id[i] = 0.0f;
    if (i < B)     out_rate[i] = 0.0f;
    if (i < B * HID) { float v = hidden[i]; h[i] = v; h_h[i] = (_Float16)v; }
}

// ------------------------------ GRU gates ----------------------------------
__global__ void gru_gates(const float* __restrict__ gi, const float* __restrict__ gh,
                          float* __restrict__ h, _Float16* __restrict__ h_h) {
    int i = blockIdx.x * blockDim.x + threadIdx.x;
    if (i >= B * HID) return;
    int b = i / HID, j = i % HID;
    const float* gib = gi + (size_t)b * H3;
    const float* ghb = gh + (size_t)b * H3;
    float r  = 1.0f / (1.0f + __expf(-(gib[j] + ghb[j])));
    float z  = 1.0f / (1.0f + __expf(-(gib[HID + j] + ghb[HID + j])));
    float n  = tanhf(gib[2 * HID + j] + r * ghb[2 * HID + j]);
    float hv = (1.0f - z) * n + z * h[i];
    h[i]   = hv;
    h_h[i] = (_Float16)hv;
}

// ----------------------- attention score reduction -------------------------
// One wave per (b,p): s = sum_j attnv[j] * tanh(q[b][j] + path_part[b,p][j])
__global__ void attn_scores(const float* __restrict__ q_part,
                            const float* __restrict__ path_part,
                            const float* __restrict__ attnv,
                            const int* __restrict__ path_len,
                            float* __restrict__ scores) {
    int gid  = blockIdx.x * blockDim.x + threadIdx.x;
    int wave = gid >> 5, lane = gid & 31;
    if (wave >= B * P) return;
    int b = wave / P, p = wave % P;
    const float* qp = q_part    + (size_t)b    * HID;
    const float* pp = path_part + (size_t)wave * HID;
    float s = 0.0f;
    for (int j = lane; j < HID; j += 32)
        s += attnv[j] * tanhf(qp[j] + pp[j]);
#pragma unroll
    for (int off = 16; off > 0; off >>= 1) s += __shfl_down(s, off, 32);
    if (lane == 0)
        scores[wave] = (p < path_len[b]) ? s : -1e9f;
}

// -------- softmax over P, weighted path sum, build rate_in = [h|weighted] ---
__global__ void softmax_weighted(const float* __restrict__ scores,
                                 const _Float16* __restrict__ path_emb,
                                 const float* __restrict__ h,
                                 float* __restrict__ out_id_t,
                                 _Float16* __restrict__ rate_in) {
    __shared__ float w[P];
    __shared__ float red[4];
    int b = blockIdx.x, tid = threadIdx.x;          // 128 threads, tid == p
    float s = scores[b * P + tid];

    float m = s;
#pragma unroll
    for (int off = 16; off > 0; off >>= 1) m = fmaxf(m, __shfl_down(m, off, 32));
    if ((tid & 31) == 0) red[tid >> 5] = m;
    __syncthreads();
    float mx = fmaxf(fmaxf(red[0], red[1]), fmaxf(red[2], red[3]));

    float e = __expf(s - mx);
    float sum = e;
#pragma unroll
    for (int off = 16; off > 0; off >>= 1) sum += __shfl_down(sum, off, 32);
    __syncthreads();
    if ((tid & 31) == 0) red[tid >> 5] = sum;
    __syncthreads();
    float tot = red[0] + red[1] + red[2] + red[3];

    float wv = e / tot;
    w[tid] = wv;
    out_id_t[b * P + tid] = wv;
    __syncthreads();

    // weighted[b][c] = sum_p w[p] * path_emb[b,p][c] ; 4 columns per thread
    int c0 = tid * 4;
    float a0 = 0.f, a1 = 0.f, a2 = 0.f, a3 = 0.f;
    const _Float16* pe = path_emb + (size_t)b * P * HID;
    for (int p = 0; p < P; ++p) {
        float wp = w[p];
        const _Float16* row = pe + (size_t)p * HID + c0;
        a0 += wp * (float)row[0]; a1 += wp * (float)row[1];
        a2 += wp * (float)row[2]; a3 += wp * (float)row[3];
    }
    _Float16* ri = rate_in + (size_t)b * (2 * HID);
    ri[HID + c0 + 0] = (_Float16)a0; ri[HID + c0 + 1] = (_Float16)a1;
    ri[HID + c0 + 2] = (_Float16)a2; ri[HID + c0 + 3] = (_Float16)a3;
#pragma unroll
    for (int k = 0; k < 4; ++k) {
        int j = tid * 4 + k;
        ri[j] = (_Float16)h[(size_t)b * HID + j];
    }
}

// ------------------------------ rate head ----------------------------------
__global__ void rate_out(const float* __restrict__ r1, const float* __restrict__ rate_w2,
                         const float* __restrict__ rate_b2, float* __restrict__ out_rate_t) {
    __shared__ float red[8];
    int b = blockIdx.x, tid = threadIdx.x;          // 256 threads
    float s = 0.0f;
    for (int j = tid; j < 2 * HID; j += 256)
        s += r1[(size_t)b * 2 * HID + j] * rate_w2[j];
#pragma unroll
    for (int off = 16; off > 0; off >>= 1) s += __shfl_down(s, off, 32);
    if ((tid & 31) == 0) red[tid >> 5] = s;
    __syncthreads();
    if (tid == 0) {
        float t = 0.0f;
#pragma unroll
        for (int i = 0; i < 8; ++i) t += red[i];
        out_rate_t[b] = 1.0f / (1.0f + __expf(-(t + rate_b2[0])));
    }
}

// ---------------------------------------------------------------------------
extern "C" void kernel_launch(void* const* d_in, const int* in_sizes, int n_in,
                              void* d_out, int out_size, void* d_ws, size_t ws_size,
                              hipStream_t stream) {
    const int*   trg_id   = (const int*)  d_in[0];
    const float* trg_rate = (const float*)d_in[1];
    const float* hidden   = (const float*)d_in[2];
    const int*   path     = (const int*)  d_in[3];
    const int*   path_len = (const int*)  d_in[4];
    const int*   d_rids   = (const int*)  d_in[5];
    const float* d_rates  = (const float*)d_in[6];
    const float* emb      = (const float*)d_in[7];
    const float* w_ih     = (const float*)d_in[8];
    const float* w_hh     = (const float*)d_in[9];
    const float* b_ih     = (const float*)d_in[10];
    const float* b_hh     = (const float*)d_in[11];
    const float* attnW_w  = (const float*)d_in[12];
    const float* attnW_b  = (const float*)d_in[13];
    const float* attnv_w  = (const float*)d_in[14];
    const float* rate_w1  = (const float*)d_in[15];
    const float* rate_b1  = (const float*)d_in[16];
    const float* rate_w2  = (const float*)d_in[17];
    const float* rate_b2  = (const float*)d_in[18];
    (void)in_sizes; (void)n_in; (void)out_size; (void)ws_size;

    float* out_id   = (float*)d_out;                  // (T,B,P)
    float* out_rate = out_id + (size_t)T * B * P;     // (T,B,1)

    // --- workspace bump allocator ---
    char* wsp = (char*)d_ws;
    auto alloc = [&](size_t bytes) -> char* {
        char* p = wsp; wsp += (bytes + 255) & ~(size_t)255; return p;
    };
    _Float16* Wx_h     = (_Float16*)alloc((size_t)H3 * HID * 2);   // w_ih cols 0..511
    _Float16* Wdest_h  = (_Float16*)alloc((size_t)H3 * HID * 2);   // w_ih cols 513..1024
    _Float16* Whh_h    = (_Float16*)alloc((size_t)H3 * HID * 2);
    _Float16* W1_h     = (_Float16*)alloc((size_t)HID * HID * 2);  // attnW cols 0..511
    _Float16* W2_h     = (_Float16*)alloc((size_t)HID * HID * 2);  // attnW cols 512..1023
    _Float16* Rw1_h    = (_Float16*)alloc((size_t)2*HID * 2*HID * 2);
    float*    w_rcol   = (float*)   alloc((size_t)H3 * 4);         // w_ih col 512
    float*    w_drcol  = (float*)   alloc((size_t)H3 * 4);         // w_ih col 1025
    float*    const_gi = (float*)   alloc((size_t)B * H3 * 4);
    _Float16* dest_h   = (_Float16*)alloc((size_t)B * HID * 2);
    _Float16* pathe_h  = (_Float16*)alloc((size_t)B * P * HID * 2);
    float*    path_pt  = (float*)   alloc((size_t)B * P * HID * 4);
    float*    h        = (float*)   alloc((size_t)B * HID * 4);
    _Float16* h_h      = (_Float16*)alloc((size_t)B * HID * 2);
    _Float16* embt_h   = (_Float16*)alloc((size_t)B * HID * 2);
    float*    gi       = (float*)   alloc((size_t)B * H3 * 4);
    float*    gh       = (float*)   alloc((size_t)B * H3 * 4);
    float*    q_part   = (float*)   alloc((size_t)B * HID * 4);
    float*    scores   = (float*)   alloc((size_t)B * P * 4);
    _Float16* rate_in  = (_Float16*)alloc((size_t)B * 2*HID * 2);
    float*    r1       = (float*)   alloc((size_t)B * 2*HID * 4);

    // --- one-time setup ---
    init_outputs_h<<<128, 256, 0, stream>>>(out_id, out_rate, h, h_h, hidden);

    extract_f16<<<(H3*HID + 255)/256, 256, 0, stream>>>(Wx_h,    w_ih,    H3, HID, RNN_LD, 0);
    extract_f16<<<(H3*HID + 255)/256, 256, 0, stream>>>(Wdest_h, w_ih,    H3, HID, RNN_LD, 513);
    extract_f16<<<(H3*HID + 255)/256, 256, 0, stream>>>(Whh_h,   w_hh,    H3, HID, HID,    0);
    extract_f16<<<(HID*HID + 255)/256, 256, 0, stream>>>(W1_h,   attnW_w, HID, HID, 2*HID, 0);
    extract_f16<<<(HID*HID + 255)/256, 256, 0, stream>>>(W2_h,   attnW_w, HID, HID, 2*HID, HID);
    extract_f16<<<(4*HID*HID + 255)/256, 256, 0, stream>>>(Rw1_h, rate_w1, 2*HID, 2*HID, 2*HID, 0);
    extract_col_f32<<<(H3 + 255)/256, 256, 0, stream>>>(w_rcol,  w_ih, H3, RNN_LD, 512);
    extract_col_f32<<<(H3 + 255)/256, 256, 0, stream>>>(w_drcol, w_ih, H3, RNN_LD, 1025);

    // const_gi = dest_emb @ Wdest^T + d_rates * w_drcol + b_ih      (once)
    gather_rows_f16<<<(B*HID + 255)/256, 256, 0, stream>>>(dest_h, emb, d_rids, B);
    gemm_wmma_f16<2><<<dim3(H3/128, B/16), 128, 0, stream>>>(
        dest_h, HID, Wdest_h, HID, const_gi, H3, B, H3, HID,
        b_ih, nullptr, d_rates, w_drcol, 0);

    // path_part = path_emb @ W2^T + attnW_b          (once, 2.1G MACs, NT=4)
    gather_path_f16<<<((long)B*P*HID + 255)/256, 256, 0, stream>>>(pathe_h, emb, path);
    gemm_wmma_f16<4><<<dim3(HID/256, (B*P)/16), 128, 0, stream>>>(
        pathe_h, HID, W2_h, HID, path_pt, HID, B*P, HID, HID,
        attnW_b, nullptr, nullptr, nullptr, 0);

    // --- sequential decode: 63 steps ---
    for (int t = 0; t < T - 1; ++t) {
        gather_rows_f16<<<(B*HID + 255)/256, 256, 0, stream>>>(embt_h, emb, trg_id + t*B, B);

        // gi = emb_t @ Wx^T + trg_rate_t * w_rcol + const_gi   (const_gi holds b_ih)
        gemm_wmma_f16<2><<<dim3(H3/128, B/16), 128, 0, stream>>>(
            embt_h, HID, Wx_h, HID, gi, H3, B, H3, HID,
            nullptr, const_gi, trg_rate + t*B, w_rcol, 0);
        // gh = h @ Whh^T + b_hh
        gemm_wmma_f16<2><<<dim3(H3/128, B/16), 128, 0, stream>>>(
            h_h, HID, Whh_h, HID, gh, H3, B, H3, HID,
            b_hh, nullptr, nullptr, nullptr, 0);
        gru_gates<<<(B*HID + 255)/256, 256, 0, stream>>>(gi, gh, h, h_h);

        // q_part = h @ W1^T   (bias folded into path_part)
        gemm_wmma_f16<2><<<dim3(HID/128, B/16), 128, 0, stream>>>(
            h_h, HID, W1_h, HID, q_part, HID, B, HID, HID,
            nullptr, nullptr, nullptr, nullptr, 0);

        attn_scores<<<(B*P*32)/256, 256, 0, stream>>>(q_part, path_pt, attnv_w, path_len, scores);
        softmax_weighted<<<B, P, 0, stream>>>(scores, pathe_h, h,
                                              out_id + (size_t)(t+1)*B*P, rate_in);

        // r1 = relu(rate_in @ rate_w1^T + rate_b1)
        gemm_wmma_f16<4><<<dim3((2*HID)/256, B/16), 128, 0, stream>>>(
            rate_in, 2*HID, Rw1_h, 2*HID, r1, 2*HID, B, 2*HID, 2*HID,
            rate_b1, nullptr, nullptr, nullptr, 1);
        rate_out<<<B, 256, 0, stream>>>(r1, rate_w2, rate_b2, out_rate + (size_t)(t+1)*B);
    }
}
